// GAT_CNN_DASE_90005334655056
// MI455X (gfx1250) — matrix operand order repaired and checked by reading, not compiled, not run on hardware
//
#include <hip/hip_runtime.h>
#include <cstdint>
#include <cstddef>

// ---------------------------------------------------------------------------
// MI455X / gfx1250 implementation of the GAT+CNN+MLP pipeline.
// All dense GEMMs (feature embed, GAT projections, 3 MLP layers) run on the
// bf16 WMMA pipe (v_wmma_f32_16x16x32_bf16). Each wave computes a 16x32
// output tile (one A fragment shared by two WMMAs). Layer-1 MLP fuses the
// emb_m[pairs]*emb_d[pairs] gather-product into the A-fragment loader,
// reading from a K-padded f32 copy of cnn_outputs (25.7MB, L2-resident)
// so all fragment loads are unguarded 128-bit vector loads.
// ---------------------------------------------------------------------------

typedef __attribute__((ext_vector_type(16))) __bf16  v16bf;
typedef __attribute__((ext_vector_type(8)))  float   v8f;
typedef __attribute__((ext_vector_type(4)))  unsigned u32x4;

#define GAT_SLOPE 0.2f
#define MLP_SLOPE 0.01f

__device__ __forceinline__ unsigned short f2bf(float f) {
  unsigned u = __float_as_uint(f);
  unsigned r = u + 0x7FFFu + ((u >> 16) & 1u);   // round-to-nearest-even
  return (unsigned short)(r >> 16);
}
__device__ __forceinline__ float bf2f(unsigned short s) {
  return __uint_as_float(((unsigned)s) << 16);
}

union FragAB {                 // POD union: 16 bf16 = 32 bytes
  v16bf v;
  u32x4 q[2];
  unsigned short u[16];
};

// product of 8 consecutive f32 from two rows -> 8 bf16 (unguarded, aligned)
__device__ __forceinline__ void gath8(const float* __restrict__ pm,
                                      const float* __restrict__ pd,
                                      unsigned short* dst) {
  float4 x0 = ((const float4*)pm)[0];
  float4 x1 = ((const float4*)pm)[1];
  float4 y0 = ((const float4*)pd)[0];
  float4 y1 = ((const float4*)pd)[1];
  dst[0] = f2bf(x0.x * y0.x); dst[1] = f2bf(x0.y * y0.y);
  dst[2] = f2bf(x0.z * y0.z); dst[3] = f2bf(x0.w * y0.w);
  dst[4] = f2bf(x1.x * y1.x); dst[5] = f2bf(x1.y * y1.y);
  dst[6] = f2bf(x1.z * y1.z); dst[7] = f2bf(x1.w * y1.w);
}

// ---------------------------------------------------------------- utilities
__global__ void fill_u32_kernel(unsigned* p, long long n, unsigned val) {
  long long i = (long long)blockIdx.x * blockDim.x + threadIdx.x;
  if (i < n) p[i] = val;
}

// dst (Mp x Kp) bf16 <- src (M x K) f32, zero padded
__global__ void cvt_pad_bf16_kernel(unsigned short* __restrict__ dst,
                                    const float* __restrict__ src,
                                    int M, int K, int Kp, long long total) {
  long long i = (long long)blockIdx.x * blockDim.x + threadIdx.x;
  if (i >= total) return;
  int m = (int)(i / Kp), k = (int)(i % Kp);
  float v = (m < M && k < K) ? src[(size_t)m * K + k] : 0.f;
  dst[i] = f2bf(v);
}

// dst (Np x Kp) bf16 <- transpose of src (K x N) f32, zero padded
__global__ void cvt_padT_bf16_kernel(unsigned short* __restrict__ dst,
                                     const float* __restrict__ src,
                                     int K, int N, int Kp, long long total) {
  long long i = (long long)blockIdx.x * blockDim.x + threadIdx.x;
  if (i >= total) return;
  int n = (int)(i / Kp), k = (int)(i % Kp);
  float v = (n < N && k < K) ? src[(size_t)k * N + n] : 0.f;
  dst[i] = f2bf(v);
}

// ------------------------------------------------------------- WMMA GEMM
// C(MxN) = A(MxKp) * Bt(NpxKp)^T, bf16 inputs, f32 accumulate.
// One wave computes a 16x32 tile: one A fragment feeds two WMMAs.
// GATHER=true: A(m,k) = cnnp[rm(m)][k] * cnnp[1500+rd(m)][k] (rows K-padded).
// Epilogue: +bias, leaky-relu, store f32 and/or bf16 (pad columns -> 0).
template <bool GATHER>
__global__ void __launch_bounds__(256)
gemm_bf16_wmma_kernel(const unsigned short* __restrict__ A, int lda,
                      const unsigned short* __restrict__ Bt, int ldb,
                      int M, int Nreal, int Kp, int ntn, int tiles,
                      const float* __restrict__ bias, float slope, int act,
                      float* __restrict__ outF, int ldoF,
                      unsigned short* __restrict__ outB, int ldoB, int orow,
                      const float* __restrict__ gcnn, int gld,
                      const int* __restrict__ posp, const int* __restrict__ negp,
                      int gRowOffD) {
  int wave = threadIdx.x >> 5;
  int lane = threadIdx.x & 31;
  int t = blockIdx.x * 8 + wave;
  if (t >= tiles) return;                 // wave-uniform: EXEC all-ones below
  int tm = t / ntn;                       // ntn = #32-wide column tiles
  int tn = t % ntn;
  int hl   = lane >> 4;                   // half-lane selector
  int rowA  = tm * 16 + (lane & 15);      // A row held by this lane
  int rowB0 = tn * 32 + (lane & 15);      // Bt rows (output cols) held
  int rowB1 = rowB0 + 16;

  const float* gM = nullptr;
  const float* gD = nullptr;
  if (GATHER) {
    int rm, rd;
    if (rowA < 40000) { rm = posp[rowA * 2 + 0]; rd = posp[rowA * 2 + 1]; }
    else { rm = negp[(rowA - 40000) * 2 + 0]; rd = negp[(rowA - 40000) * 2 + 1]; }
    gM = gcnn + (size_t)rm * gld;
    gD = gcnn + (size_t)(gRowOffD + rd) * gld;
  }

  v8f acc0 = {};
  v8f acc1 = {};
#pragma unroll 2
  for (int kk = 0; kk < Kp; kk += 32) {
    FragAB a, b0, b1;
    if (!GATHER) {
      // A 16x32 bf16 layout: elems 0..7 -> K = kk + hl*8 + i,
      //                      elems 8..15 -> K = kk + 16 + hl*8 + i
      const u32x4* pa = (const u32x4*)(A + (size_t)rowA * lda + kk + hl * 8);
      a.q[0] = pa[0];
      a.q[1] = pa[2];                     // +16 bf16 elements
    } else {
      int kb = kk + hl * 8;               // rows padded to Kp: no guards
      gath8(gM + kb,      gD + kb,      &a.u[0]);
      gath8(gM + kb + 16, gD + kb + 16, &a.u[8]);
    }
    // B 32x16 bf16 layout: lane holds col n = lane&15, elems 0..7 -> K = kk + hl*8 + i,
    //                      elems 8..15 -> K = kk + 16 + hl*8 + i
    const unsigned short* pb0 = Bt + (size_t)rowB0 * ldb + kk + hl * 8;
    const unsigned short* pb1 = Bt + (size_t)rowB1 * ldb + kk + hl * 8;
    b0.q[0] = ((const u32x4*)pb0)[0];
    b0.q[1] = ((const u32x4*)pb0)[2];
    b1.q[0] = ((const u32x4*)pb1)[0];
    b1.q[1] = ((const u32x4*)pb1)[2];
    if (kk + 32 < Kp) {                   // CDNA5 global_prefetch_b8
      __builtin_prefetch((const void*)(pb0 + 32), 0, 0);
      __builtin_prefetch((const void*)(pb1 + 32), 0, 0);
    }
    acc0 = __builtin_amdgcn_wmma_f32_16x16x32_bf16(
        false, a.v, false, b0.v, (short)0, acc0, false, false);
    acc1 = __builtin_amdgcn_wmma_f32_16x16x32_bf16(
        false, a.v, false, b1.v, (short)0, acc1, false, false);
  }

#pragma unroll
  for (int half = 0; half < 2; ++half) {
    int n = tn * 32 + half * 16 + (lane & 15);
#pragma unroll
    for (int v = 0; v < 8; ++v) {
      int m = tm * 16 + hl * 8 + v;       // C/D layout: lanes>=16 hold M>=8
      if (m >= M) continue;
      float val = half ? acc1[v] : acc0[v];
      bool realn = (n < Nreal);
      if (realn) {
        if (bias) val += bias[n];
        if (act)  val = (val > 0.f) ? val : slope * val;
      } else {
        val = 0.f;                        // zero K/N padding for next layer
      }
      if (outF && realn) outF[(size_t)(orow + m) * ldoF + n] = val;
      if (outB && n < ldoB) outB[(size_t)(orow + m) * ldoB + n] = f2bf(val);
    }
  }
}

// ------------------------------------------------------------- GAT kernels
__device__ __forceinline__ void atomicMaxF(float* addr, float val) {
  unsigned* ua = (unsigned*)addr;
  unsigned old = *ua;
  while (true) {
    float f = __uint_as_float(old);
    if (f >= val) break;
    unsigned assumed = old;
    old = atomicCAS(ua, assumed, __float_as_uint(val));
    if (old == assumed) break;
  }
}

__global__ void build_edges_kernel(const int* __restrict__ eidx, int E, int N,
                                   int* __restrict__ src, int* __restrict__ dst) {
  int i = blockIdx.x * blockDim.x + threadIdx.x;
  if (i >= E + N) return;
  int s, d;
  if (i < E) { s = eidx[i]; d = eidx[E + i]; }
  else       { s = i - E;  d = i - E; }
  src[i] = s; dst[i] = d;
}

// one wave per (edge, head): e[eh] = sum_c lrelu(xl[s]+xr[d]) * att ; seg max
__global__ void __launch_bounds__(256)
edge_score_kernel(const int* __restrict__ src, const int* __restrict__ dst,
                  const float* __restrict__ xl, const float* __restrict__ xr,
                  const float* __restrict__ att, int EN,
                  float* __restrict__ ebuf, float* __restrict__ emax) {
  int w = blockIdx.x * 8 + (threadIdx.x >> 5);
  if (w >= EN * 8) return;
  int e = w >> 3, h = w & 7;
  int lane = threadIdx.x & 31;
  int s = src[e], d = dst[e];
  const float4* pl = (const float4*)(xl + (size_t)s * 1024 + h * 128);
  const float4* pr = (const float4*)(xr + (size_t)d * 1024 + h * 128);
  const float4* pw = (const float4*)(att + h * 128);
  float4 a = pl[lane], b = pr[lane], wv = pw[lane];
  float z0 = a.x + b.x; z0 = z0 > 0.f ? z0 : GAT_SLOPE * z0;
  float z1 = a.y + b.y; z1 = z1 > 0.f ? z1 : GAT_SLOPE * z1;
  float z2 = a.z + b.z; z2 = z2 > 0.f ? z2 : GAT_SLOPE * z2;
  float z3 = a.w + b.w; z3 = z3 > 0.f ? z3 : GAT_SLOPE * z3;
  float acc = z0 * wv.x + z1 * wv.y + z2 * wv.z + z3 * wv.w;
  for (int o = 16; o > 0; o >>= 1) acc += __shfl_xor(acc, o, 32);
  if (lane == 0) {
    ebuf[e * 8 + h] = acc;
    atomicMaxF(&emax[d * 8 + h], acc);
  }
}

__global__ void edge_softmax_kernel(const int* __restrict__ dst,
                                    const float* __restrict__ ebuf,
                                    const float* __restrict__ emax,
                                    float* __restrict__ exbuf,
                                    float* __restrict__ den, int EN) {
  int i = blockIdx.x * blockDim.x + threadIdx.x;
  if (i >= EN * 8) return;
  int e = i >> 3, h = i & 7;
  int d = dst[e];
  float v = expf(ebuf[i] - emax[d * 8 + h]);
  exbuf[i] = v;
  atomicAdd(&den[d * 8 + h], v);
}

// out[dst] += alpha * xl[src]  (one wave per (edge, head))
__global__ void __launch_bounds__(256)
edge_aggregate_kernel(const int* __restrict__ src, const int* __restrict__ dst,
                      const float* __restrict__ xl,
                      const float* __restrict__ exbuf,
                      const float* __restrict__ den,
                      float* __restrict__ outacc, int EN) {
  int w = blockIdx.x * 8 + (threadIdx.x >> 5);
  if (w >= EN * 8) return;
  int e = w >> 3, h = w & 7;
  int lane = threadIdx.x & 31;
  int s = src[e], d = dst[e];
  float alpha = exbuf[e * 8 + h] / den[d * 8 + h];
  const float4* pl = (const float4*)(xl + (size_t)s * 1024 + h * 128);
  float4 xv = pl[lane];
  float* po = outacc + (size_t)d * 1024 + h * 128 + lane * 4;
  atomicAdd(po + 0, alpha * xv.x);
  atomicAdd(po + 1, alpha * xv.y);
  atomicAdd(po + 2, alpha * xv.z);
  atomicAdd(po + 3, alpha * xv.w);
}

// ------------------------------------------------------------- CNN kernel
// one block per node; res (8x128) staged in LDS; 4 conv widths concatenated.
// Writes both the exact cnn_outputs (ld 2778, to d_out) and a K-padded
// f32 copy (ld 2784, pad cols = 0) used by the fused layer-1 gather.
__global__ void __launch_bounds__(256)
conv_kernel(const float* __restrict__ gatout, const float* __restrict__ gbias,
            const float* __restrict__ cw1,  const float* __restrict__ cb1,
            const float* __restrict__ cw4,  const float* __restrict__ cb4,
            const float* __restrict__ cw16, const float* __restrict__ cb16,
            const float* __restrict__ cw32, const float* __restrict__ cb32,
            float* __restrict__ cnn, float* __restrict__ cnnp) {
  __shared__ float sx[1024];
  __shared__ float sw[2544];
  __shared__ float sb[24];
  int n = blockIdx.x, tid = threadIdx.x;
  for (int i = tid; i < 1024; i += 256) sx[i] = gatout[(size_t)n * 1024 + i] + gbias[i];
  for (int i = tid; i < 48;   i += 256) sw[i]        = cw1[i];
  for (int i = tid; i < 192;  i += 256) sw[48 + i]   = cw4[i];
  for (int i = tid; i < 768;  i += 256) sw[240 + i]  = cw16[i];
  for (int i = tid; i < 1536; i += 256) sw[1008 + i] = cw32[i];
  if (tid < 6) { sb[tid] = cb1[tid]; sb[6+tid] = cb4[tid];
                 sb[12+tid] = cb16[tid]; sb[18+tid] = cb32[tid]; }
  __syncthreads();
  for (int j = tid; j < 2784; j += 256) {
    if (j >= 2778) { cnnp[(size_t)n * 2784 + j] = 0.f; continue; }
    int loc, kw, W, woff, boff;
    if (j < 768)       { loc = j;        kw = 1;  W = 128; woff = 0;    boff = 0; }
    else if (j < 1518) { loc = j - 768;  kw = 4;  W = 125; woff = 48;   boff = 6; }
    else if (j < 2196) { loc = j - 1518; kw = 16; W = 113; woff = 240;  boff = 12; }
    else               { loc = j - 2196; kw = 32; W = 97;  woff = 1008; boff = 18; }
    int o = loc / W, wp = loc % W;
    float acc = sb[boff + o];
    const float* wk = &sw[woff + o * 8 * kw];
    for (int kh = 0; kh < 8; ++kh) {
      const float* xrow = &sx[kh * 128 + wp];
      const float* wrow = &wk[kh * kw];
      for (int q = 0; q < kw; ++q) acc += xrow[q] * wrow[q];
    }
    float r = acc > 0.f ? acc : 0.f;
    cnn [(size_t)n * 2778 + j] = r;
    cnnp[(size_t)n * 2784 + j] = r;
  }
}

// ------------------------------------------------------------- final layer
__global__ void __launch_bounds__(256)
mlp_final_kernel(const unsigned short* __restrict__ h3,
                 const float* __restrict__ mw4,
                 float* __restrict__ pred, float* __restrict__ labels) {
  int m = blockIdx.x * 8 + (threadIdx.x >> 5);
  if (m >= 80000) return;
  int lane = threadIdx.x & 31;
  float acc = 0.f;
  for (int k = lane; k < 463; k += 32)
    acc += bf2f(h3[(size_t)m * 464 + k]) * mw4[k];
  for (int o = 16; o > 0; o >>= 1) acc += __shfl_xor(acc, o, 32);
  if (lane == 0) {
    pred[m] = 1.f / (1.f + expf(-acc));
    labels[m] = (m < 40000) ? 1.f : 0.f;
  }
}

// ---------------------------------------------------------------------------
extern "C" void kernel_launch(void* const* d_in, const int* in_sizes, int n_in,
                              void* d_out, int out_size, void* d_ws, size_t ws_size,
                              hipStream_t stream) {
  (void)in_sizes; (void)n_in; (void)out_size; (void)ws_size;

  const float* mic   = (const float*)d_in[0];
  const float* dis   = (const float*)d_in[1];
  const int*   eidx  = (const int*)d_in[2];
  const int*   posp  = (const int*)d_in[3];
  const int*   negp  = (const int*)d_in[4];
  const float* W_mic = (const float*)d_in[6];
  const float* W_dis = (const float*)d_in[7];
  const float* gWl   = (const float*)d_in[8];
  const float* gbl   = (const float*)d_in[9];
  const float* gWr   = (const float*)d_in[10];
  const float* gbr   = (const float*)d_in[11];
  const float* gatt  = (const float*)d_in[12];
  const float* gbias = (const float*)d_in[13];
  const float* cw1   = (const float*)d_in[14];
  const float* cb1   = (const float*)d_in[15];
  const float* cw4   = (const float*)d_in[16];
  const float* cb4   = (const float*)d_in[17];
  const float* cw16  = (const float*)d_in[18];
  const float* cb16  = (const float*)d_in[19];
  const float* cw32  = (const float*)d_in[20];
  const float* cb32  = (const float*)d_in[21];
  const float* mw1   = (const float*)d_in[22];
  const float* mb1   = (const float*)d_in[23];
  const float* mw2   = (const float*)d_in[24];
  const float* mb2   = (const float*)d_in[25];
  const float* mw3   = (const float*)d_in[26];
  const float* mb3   = (const float*)d_in[27];
  const float* mw4   = (const float*)d_in[28];

  const int E = 80000, NN = 2300, EN = E + NN;

  float* pred   = (float*)d_out;
  float* labels = (float*)d_out + 80000;
  float* cnn    = (float*)d_out + 160000;     // (2300 x 2778) f32

  // ---- workspace layout (bump allocator, 256B aligned, ~490 MB total)
  char* ws = (char*)d_ws;
  size_t off = 0;
  auto bump = [&](size_t bytes) -> char* {
    char* p = ws + off;
    off += (bytes + 255) & ~(size_t)255;
    return p;
  };
  unsigned short* mic_bf = (unsigned short*)bump(1504ull * 1504 * 2);
  unsigned short* wmicT  = (unsigned short*)bump(128ull * 1504 * 2);
  unsigned short* dis_bf = (unsigned short*)bump(800ull * 800 * 2);
  unsigned short* wdisT  = (unsigned short*)bump(128ull * 800 * 2);
  unsigned short* h_bf   = (unsigned short*)bump(2304ull * 128 * 2);
  unsigned short* wlT    = (unsigned short*)bump(1024ull * 128 * 2);
  unsigned short* wrT    = (unsigned short*)bump(1024ull * 128 * 2);
  float* xl      = (float*)bump(2304ull * 1024 * 4);
  float* xr      = (float*)bump(2304ull * 1024 * 4);
  int*   srcA    = (int*)bump(82304ull * 4);
  int*   dstA    = (int*)bump(82304ull * 4);
  float* ebuf    = (float*)bump(82304ull * 8 * 4);
  float* exbuf   = (float*)bump(82304ull * 8 * 4);
  float* emax    = (float*)bump(2304ull * 8 * 4);
  float* den     = (float*)bump(2304ull * 8 * 4);
  float* gatout  = (float*)bump(2304ull * 1024 * 4);
  float* cnnp    = (float*)bump(2300ull * 2784 * 4);     // K-padded cnn copy
  unsigned short* w1T = (unsigned short*)bump(1408ull * 2784 * 2);
  unsigned short* w2T = (unsigned short*)bump(704ull * 1408 * 2);
  unsigned short* w3T = (unsigned short*)bump(480ull * 704 * 2);  // Np=480
  unsigned short* h1  = (unsigned short*)bump(80000ull * 1408 * 2);
  unsigned short* h2  = (unsigned short*)bump(80000ull * 704 * 2);
  unsigned short* h3  = (unsigned short*)bump(80000ull * 464 * 2);

  auto fill = [&](void* p, long long words, unsigned val) {
    fill_u32_kernel<<<(unsigned)((words + 255) / 256), 256, 0, stream>>>(
        (unsigned*)p, words, val);
  };
  auto cvtP = [&](unsigned short* dst, const float* src, int M, int K, int Mp, int Kp) {
    long long total = (long long)Mp * Kp;
    cvt_pad_bf16_kernel<<<(unsigned)((total + 255) / 256), 256, 0, stream>>>(
        dst, src, M, K, Kp, total);
  };
  auto cvtT = [&](unsigned short* dst, const float* src, int K, int N, int Kp, int Np) {
    long long total = (long long)Np * Kp;
    cvt_padT_bf16_kernel<<<(unsigned)((total + 255) / 256), 256, 0, stream>>>(
        dst, src, K, N, Kp, total);
  };
  // Np must be a multiple of 32 (16x32 wave tiles)
  auto gemm = [&](const unsigned short* A, int lda, const unsigned short* Bt, int ldb,
                  int M, int Nreal, int Kp, int Np,
                  const float* bias, float slope, int act,
                  float* outF, int ldoF, unsigned short* outB, int ldoB, int orow,
                  const float* gsrc, const int* pp, const int* np, int useGather) {
    int ntm = (M + 15) / 16;
    int ntn = Np / 32;
    int tiles = ntm * ntn;
    unsigned blocks = (unsigned)((tiles + 7) / 8);
    if (useGather) {
      gemm_bf16_wmma_kernel<true><<<blocks, 256, 0, stream>>>(
          A, lda, Bt, ldb, M, Nreal, Kp, ntn, tiles,
          bias, slope, act, outF, ldoF, outB, ldoB, orow,
          gsrc, /*gld=*/2784, pp, np, /*gRowOffD=*/1500);
    } else {
      gemm_bf16_wmma_kernel<false><<<blocks, 256, 0, stream>>>(
          A, lda, Bt, ldb, M, Nreal, Kp, ntn, tiles,
          bias, slope, act, outF, ldoF, outB, ldoB, orow,
          nullptr, 0, nullptr, nullptr, 0);
    }
  };

  // ---- per-call init (graph-replay safe: everything rewritten every call)
  fill(h_bf,   2304ll * 128 / 2, 0u);
  fill(emax,   2304ll * 8,       0xFF800000u);   // -inf
  fill(den,    2304ll * 8,       0u);
  fill(gatout, 2304ll * 1024,    0u);

  // ---- bf16 conversions (A padded row-major, B padded transposed)
  cvtP(mic_bf, mic, 1500, 1500, 1504, 1504);
  cvtP(dis_bf, dis, 800, 800, 800, 800);
  cvtT(wmicT, W_mic, 1500, 128, 1504, 128);
  cvtT(wdisT, W_dis, 800, 128, 800, 128);
  cvtT(wlT, gWl, 128, 1024, 128, 1024);
  cvtT(wrT, gWr, 128, 1024, 128, 1024);
  cvtT(w1T, mw1, 2778, 1389, 2784, 1408);
  cvtT(w2T, mw2, 1389, 694, 1408, 704);
  cvtT(w3T, mw3, 694, 463, 704, 480);

  // ---- h = [mic@W_mic ; dis@W_dis]  (stored bf16 for next GEMM)
  gemm(mic_bf, 1504, wmicT, 1504, 1500, 128, 1504, 128,
       nullptr, 0.f, 0, nullptr, 0, h_bf, 128, 0, nullptr, nullptr, nullptr, 0);
  gemm(dis_bf, 800, wdisT, 800, 800, 128, 800, 128,
       nullptr, 0.f, 0, nullptr, 0, h_bf, 128, 1500, nullptr, nullptr, nullptr, 0);

  // ---- xl = h@Wl + bl ; xr = h@Wr + br
  gemm(h_bf, 128, wlT, 128, NN, 1024, 128, 1024,
       gbl, 0.f, 0, xl, 1024, nullptr, 0, 0, nullptr, nullptr, nullptr, 0);
  gemm(h_bf, 128, wrT, 128, NN, 1024, 128, 1024,
       gbr, 0.f, 0, xr, 1024, nullptr, 0, 0, nullptr, nullptr, nullptr, 0);

  // ---- GAT edge softmax + aggregation
  build_edges_kernel<<<(EN + 255) / 256, 256, 0, stream>>>(eidx, E, NN, srcA, dstA);
  edge_score_kernel<<<EN, 256, 0, stream>>>(srcA, dstA, xl, xr, gatt, EN, ebuf, emax);
  edge_softmax_kernel<<<(EN * 8 + 255) / 256, 256, 0, stream>>>(
      dstA, ebuf, emax, exbuf, den, EN);
  edge_aggregate_kernel<<<EN, 256, 0, stream>>>(srcA, dstA, xl, exbuf, den, gatout, EN);

  // ---- CNN head -> cnn_outputs (d_out) + K-padded copy for the gather
  conv_kernel<<<NN, 256, 0, stream>>>(gatout, gbias, cw1, cb1, cw4, cb4,
                                      cw16, cb16, cw32, cb32, cnn, cnnp);

  // ---- MLP (dominant 822 GFLOP): bf16 WMMA GEMMs, layer-1 gather fused
  gemm(nullptr, 0, w1T, 2784, 80000, 1389, 2784, 1408,
       mb1, MLP_SLOPE, 1, nullptr, 0, h1, 1408, 0, cnnp, posp, negp, 1);
  gemm(h1, 1408, w2T, 1408, 80000, 694, 1408, 704,
       mb2, MLP_SLOPE, 1, nullptr, 0, h2, 704, 0, nullptr, nullptr, nullptr, 0);
  gemm(h2, 704, w3T, 704, 80000, 463, 704, 480,
       mb3, MLP_SLOPE, 1, nullptr, 0, h3, 464, 0, nullptr, nullptr, nullptr, 0);

  // ---- final dot + sigmoid + labels
  mlp_final_kernel<<<10000, 256, 0, stream>>>(h3, mw4, pred, labels);
}
